// MultiHeadAttention_17712445128719
// MI455X (gfx1250) — matrix operand order repaired and checked by reading, not compiled, not run on hardware
//
#include <hip/hip_runtime.h>

// ---------------------------------------------------------------------------
// MultiHeadAttention on gfx1250 (MI455X, CDNA5, wave32, WMMA f16->f32)
//
//  * pad(-1e19 on masked rows OR cols) added BEFORE 1/sqrt(dk) scaling
//  * softmax over the QUERY axis (column softmax): two passes over QK^T:
//      pass A: per-column online (m,Z); pass B: P=exp(s-m[k])/Z[k], P@V.
//  * K/V (pass B) and Q (pass A) tiles are shared by all 8 waves of a block
//    -> staged into LDS with double-buffered GLOBAL_LOAD_ASYNC_TO_LDS_B128
//       (ASYNCcnt / s_wait_asynccnt), cutting redundant loads 8x.
// ---------------------------------------------------------------------------

#define B_    2
#define S_    2048
#define D_    1024
#define H_    16
#define DK_   64
#define NM_   64
#define MROWS (B_ * S_)          // 4096
#define NCOL  (H_ * DK_)         // 1024

typedef _Float16 h16;
typedef __attribute__((ext_vector_type(16))) _Float16 v16h;
typedef __attribute__((ext_vector_type(8)))  _Float16 h8;
typedef __attribute__((ext_vector_type(8)))  float    v8f;

#define WMMA_F16(a, b, c) \
  __builtin_amdgcn_wmma_f32_16x16x32_f16(false, (a), false, (b), (short)0, (c), false, false)

#define ASYNC_CP_B128(ldsoff, gaddr) \
  asm volatile("global_load_async_to_lds_b128 %0, %1, off" :: "v"(ldsoff), "v"(gaddr) : "memory")
#define WAIT_ASYNC0() asm volatile("s_wait_asynccnt 0x0" ::: "memory")

// LDS byte offset of a __shared__ object = low 32 bits of its flat address
// (flat->LDS mapping truncates to addr[31:0]).
__device__ __forceinline__ unsigned lds_off(const void* p) {
  return (unsigned)(unsigned long long)(uintptr_t)p;
}

// ---- Fragment loaders (ISA 7.12.2 16-bit layouts, wave32) -----------------
// Both A and B fragments are two contiguous 16-byte runs per lane.

// A (16x32, MxK): lane l -> row M=l&15; elements [0..7]=K(8*half..),
// [8..15]=K(16+8*half..) -> two b128 loads.
__device__ __forceinline__ v16h load_A_frag(const h16* __restrict__ A, int lda, int lane) {
  const int half = lane >> 4;
  const h16* row = A + (size_t)(lane & 15) * lda + half * 8;
  h8 lo = *(const h8*)(row);
  h8 hi = *(const h8*)(row + 16);
  v16h a;
#pragma unroll
  for (int i = 0; i < 8; ++i) { a[i] = lo[i]; a[8 + i] = hi[i]; }
  return a;
}

// B (32x16, KxN) from TRANSPOSED array Bt[N][K] (row-major, stride ldb):
// lane l -> col N=l&15; elements [0..7]=K(8*half..), [8..15]=K(16+8*half..).
__device__ __forceinline__ v16h load_Bt_frag(const h16* __restrict__ Bt, int ldb, int lane) {
  const h16* row = Bt + (size_t)(lane & 15) * ldb + ((lane >> 4) * 8);
  h8 lo = *(const h8*)(row);
  h8 hi = *(const h8*)(row + 16);
  v16h b;
#pragma unroll
  for (int i = 0; i < 8; ++i) { b[i] = lo[i]; b[8 + i] = hi[i]; }
  return b;
}

// ---- Small prep kernels ---------------------------------------------------

__global__ void k_f32_to_f16(const float* __restrict__ in, h16* __restrict__ out, int n) {
  int i = blockIdx.x * blockDim.x + threadIdx.x;
  if (i < n) out[i] = (h16)in[i];
}

// CWt[n][d] = sum_k Wcomb[d][k] * W[h][k][e]   (n = h*64+e), transposed f16
__global__ void k_combine_w(const float* __restrict__ Wcomb, const float* __restrict__ W,
                            h16* __restrict__ CWt) {
  int idx = blockIdx.x * blockDim.x + threadIdx.x;
  if (idx >= D_ * NCOL) return;
  const int d = idx & (D_ - 1);
  const int n = idx >> 10;
  const int h = n >> 6, e = n & 63;
  const float* wc = Wcomb + (size_t)d * DK_;
  const float* wh = W + (size_t)h * DK_ * DK_ + e;
  float s = 0.f;
#pragma unroll 8
  for (int k = 0; k < DK_; ++k) s += wc[k] * wh[k * DK_];
  CWt[idx] = (h16)s;
}

__global__ void k_transpose_w(const float* __restrict__ W, h16* __restrict__ Wt) {
  int idx = blockIdx.x * blockDim.x + threadIdx.x;
  if (idx >= NCOL * D_) return;
  const int n = idx >> 10, k = idx & (D_ - 1);
  Wt[idx] = (h16)W[(size_t)k * D_ + n];
}

__global__ void k_zero_flags(unsigned char* f, int n) {
  int i = blockIdx.x * blockDim.x + threadIdx.x;
  if (i < n) f[i] = 0;
}

__global__ void k_set_flags(const int* __restrict__ mi, unsigned char* __restrict__ f) {
  int i = blockIdx.x * blockDim.x + threadIdx.x;
  if (i >= B_ * NM_) return;
  const int b = i / NM_;
  f[b * S_ + mi[i]] = 1;
}

// V [b][s][h][e] -> Vt [b][h][e][s]
__global__ void k_transpose_v(const h16* __restrict__ Vh, h16* __restrict__ Vt) {
  int idx = blockIdx.x * blockDim.x + threadIdx.x;
  if (idx >= B_ * S_ * NCOL) return;
  const int e = idx & 63;
  const int h = (idx >> 6) & (H_ - 1);
  const int s = (idx >> 10) & (S_ - 1);
  const int b = idx >> 21;
  Vt[(((size_t)(b * H_ + h) * DK_) + e) * S_ + s] = Vh[idx];
}

// ---- Generic WMMA GEMM: one wave computes a 32x64 tile --------------------

__global__ void k_gemm16(const h16* __restrict__ A, const h16* __restrict__ Bt,
                         const float* __restrict__ bias,
                         float* __restrict__ Cf, h16* __restrict__ Ch,
                         int M, int N, int K) {
  const int wave = threadIdx.x >> 5, lane = threadIdx.x & 31;
  const int m0 = (blockIdx.x * 8 + wave) << 5;
  const int n0 = blockIdx.y << 6;
  if (m0 >= M) return;

  v8f acc[2][4] = {};
  for (int k0 = 0; k0 < K; k0 += 32) {
    v16h a0 = load_A_frag(A + (size_t)m0 * K + k0, K, lane);
    v16h a1 = load_A_frag(A + (size_t)(m0 + 16) * K + k0, K, lane);
#pragma unroll
    for (int t = 0; t < 4; ++t) {
      v16h b = load_Bt_frag(Bt + (size_t)(n0 + 16 * t) * K + k0, K, lane);
      acc[0][t] = WMMA_F16(a0, b, acc[0][t]);
      acc[1][t] = WMMA_F16(a1, b, acc[1][t]);
    }
  }
  const int nl = lane & 15, half = lane >> 4;
#pragma unroll
  for (int u = 0; u < 2; ++u)
#pragma unroll
    for (int t = 0; t < 4; ++t) {
      const int col = n0 + 16 * t + nl;
      const float bv = bias ? bias[col] : 0.f;
#pragma unroll
      for (int r = 0; r < 8; ++r) {
        const size_t idx = (size_t)(m0 + 16 * u + r + 8 * half) * N + col;
        const float val = acc[u][t][r] + bv;
        if (Cf) Cf[idx] = val;
        else    Ch[idx] = (h16)val;
      }
    }
}

// ---- Pass A: per-column (softmax over q) stats m[k], Z[k] -----------------
// 8 waves share the q loop -> Q tiles async-staged into LDS, double-buffered.

__global__ void k_attn_colstats(const h16* __restrict__ Q, const h16* __restrict__ Kc,
                                const unsigned char* __restrict__ flags,
                                float* __restrict__ mOut, float* __restrict__ zOut) {
  const float scale = 0.125f;  // 1/sqrt(64)
  __shared__ h16 Qbuf[2][16][64];  // 2 x 2KB

  const int tid = threadIdx.x;
  const int wave = tid >> 5, lane = tid & 31;
  const int bh = blockIdx.x;
  const int b = bh >> 4, h = bh & 15;
  const int kt = (blockIdx.y * 8 + wave) << 4;

  const h16* Qh = Q + (size_t)b * S_ * NCOL + h * DK_;
  const h16* Kh = Kc + (size_t)b * S_ * NCOL + h * DK_;
  const unsigned char* fl = flags + b * S_;
  const unsigned long long QB = (unsigned long long)(uintptr_t)Qh;

  const int nl = lane & 15, half = lane >> 4;
  const int kcol = kt + nl;
  const float fkpad = fl[kcol] ? -1e19f : 0.0f;

  // K-tile B fragments reused for every q step
  const v16h bk0 = load_Bt_frag(Kh + (size_t)kt * NCOL + 0, NCOL, lane);
  const v16h bk1 = load_Bt_frag(Kh + (size_t)kt * NCOL + 32, NCOL, lane);

  // stage a 16x64 Q tile (2KB = 128 x b128): waves 0..3, thread t -> xfer t
  auto stageQ = [&](int buf, int qtn) {
    if (wave < 4) {
      const unsigned long long g =
          QB + (unsigned long long)(qtn + (tid >> 3)) * (NCOL * 2) + (tid & 7) * 16;
      const unsigned l = lds_off(&Qbuf[buf][tid >> 3][(tid & 7) * 8]);
      ASYNC_CP_B128(l, g);
    }
  };
  stageQ(0, 0);

  float mrun = -3.0e38f, z = 0.0f;
  for (int qt = 0; qt < S_; qt += 16) {
    const int buf = (qt >> 4) & 1;
    WAIT_ASYNC0();
    __syncthreads();
    if (qt + 16 < S_) stageQ(buf ^ 1, qt + 16);

    v16h a0 = load_A_frag(&Qbuf[buf][0][0], 64, lane);
    v16h a1 = load_A_frag(&Qbuf[buf][0][32], 64, lane);
    v8f c = {};
    c = WMMA_F16(a0, bk0, c);
    c = WMMA_F16(a1, bk1, c);

    const unsigned long long fq = *(const unsigned long long*)(fl + qt + 8 * half);
    float vals[8];
    float vmax = -3.0e38f;
#pragma unroll
    for (int r = 0; r < 8; ++r) {
      const float pad = (((fq >> (8 * r)) & 0xffull) != 0) ? -1e19f : fkpad;
      const float v = (c[r] + pad) * scale;  // mask BEFORE scale (reference quirk)
      vals[r] = v;
      vmax = fmaxf(vmax, v);
    }
    vmax = fmaxf(vmax, __shfl_xor(vmax, 16, 32));  // combine q rows 0-7 / 8-15
    const float mnew = fmaxf(mrun, vmax);
    const float fac = __expf(mrun - mnew);
    float sl = 0.0f;
#pragma unroll
    for (int r = 0; r < 8; ++r) sl += __expf(vals[r] - mnew);
    sl += __shfl_xor(sl, 16, 32);
    z = z * fac + sl;
    mrun = mnew;
  }
  if (half == 0) {
    mOut[(size_t)bh * S_ + kcol] = mrun;
    zOut[(size_t)bh * S_ + kcol] = z;
  }
}

// ---- Pass B: heads[q,:] = sum_k exp(s[q,k]-m[k])/Z[k] * V[k,:] ------------
// 8 waves share (b,h) and the k loop -> K and Vt tiles async-staged into LDS.

__global__ void k_attn_pv(const h16* __restrict__ Q, const h16* __restrict__ Kc,
                          const h16* __restrict__ Vt,
                          const float* __restrict__ mIn, const float* __restrict__ zIn,
                          const unsigned char* __restrict__ flags,
                          h16* __restrict__ heads) {
  const float scale = 0.125f;
  __shared__ h16 Kbuf[2][32][64];  // 2 x 4KB, K rows kt..kt+31
  __shared__ h16 Vbuf[2][64][32];  // 2 x 4KB, Vt rows e=0..63, cols kt..kt+31
  __shared__ h16 P[8][16][32];     // per-wave P tile (C-layout -> A-layout)

  const int tid = threadIdx.x;
  const int wave = tid >> 5, lane = tid & 31;
  const int bh = blockIdx.x;
  const int b = bh >> 4, h = bh & 15;
  const int qt = (blockIdx.y * 8 + wave) << 4;

  const h16* Qh = Q + (size_t)b * S_ * NCOL + h * DK_;
  const h16* Kh = Kc + (size_t)b * S_ * NCOL + h * DK_;
  const h16* Vth = Vt + (size_t)bh * DK_ * S_;
  const unsigned char* fl = flags + b * S_;
  const float* mh = mIn + (size_t)bh * S_;
  const float* zh = zIn + (size_t)bh * S_;
  const unsigned long long KB = (unsigned long long)(uintptr_t)Kh;
  const unsigned long long VB = (unsigned long long)(uintptr_t)Vth;

  const int nl = lane & 15, half = lane >> 4;

  // Q A-fragments reused across the whole k loop
  const v16h aq0 = load_A_frag(Qh + (size_t)qt * NCOL + 0, NCOL, lane);
  const v16h aq1 = load_A_frag(Qh + (size_t)qt * NCOL + 32, NCOL, lane);
  const unsigned long long fqw = *(const unsigned long long*)(fl + qt + 8 * half);

  // stage K tile (32x128B) + Vt tile (64x64B): 512 x b128, thread t -> xfer t
  auto stageKV = [&](int buf, int ktn) {
    const unsigned long long gk =
        KB + (unsigned long long)(ktn + (tid >> 3)) * (NCOL * 2) + (tid & 7) * 16;
    const unsigned lk = lds_off(&Kbuf[buf][tid >> 3][(tid & 7) * 8]);
    ASYNC_CP_B128(lk, gk);
    const unsigned long long gv =
        VB + (unsigned long long)(tid >> 2) * (S_ * 2) + (unsigned long long)ktn * 2 + (tid & 3) * 16;
    const unsigned lv = lds_off(&Vbuf[buf][tid >> 2][(tid & 3) * 8]);
    ASYNC_CP_B128(lv, gv);
  };
  stageKV(0, 0);

  v8f acc[4] = {};
  for (int kt = 0; kt < S_; kt += 32) {
    const int buf = (kt >> 5) & 1;
    WAIT_ASYNC0();
    __syncthreads();
    if (kt + 32 < S_) stageKV(buf ^ 1, kt + 32);

#pragma unroll
    for (int sub = 0; sub < 2; ++sub) {
      v16h bk0 = load_Bt_frag(&Kbuf[buf][sub * 16][0], 64, lane);
      v16h bk1 = load_Bt_frag(&Kbuf[buf][sub * 16][32], 64, lane);
      v8f c = {};
      c = WMMA_F16(aq0, bk0, c);
      c = WMMA_F16(aq1, bk1, c);

      const int kcol = kt + sub * 16 + nl;
      const float fkpad = fl[kcol] ? -1e19f : 0.0f;
      const float mk = mh[kcol];
      const float rz = 1.0f / zh[kcol];
#pragma unroll
      for (int r = 0; r < 8; ++r) {
        const float pad = (((fqw >> (8 * r)) & 0xffull) != 0) ? -1e19f : fkpad;
        const float p = __expf((c[r] + pad) * scale - mk) * rz;
        P[wave][r + 8 * half][sub * 16 + nl] = (h16)p;  // row q, col k
      }
    }
    // Re-stripe P (C layout) into an A fragment; wave-local DS ops are in-order.
    v16h ap = load_A_frag(&P[wave][0][0], 32, lane);
#pragma unroll
    for (int t = 0; t < 4; ++t) {
      v16h bv = load_Bt_frag(&Vbuf[buf][t * 16][0], 32, lane);
      acc[t] = WMMA_F16(ap, bv, acc[t]);
    }
  }

  // heads [b*S+q][h*64+e] f16 (this IS the concat layout)
  h16* out = heads + (size_t)(b * S_ + qt) * NCOL + h * DK_;
#pragma unroll
  for (int t = 0; t < 4; ++t)
#pragma unroll
    for (int r = 0; r < 8; ++r)
      out[(size_t)(r + 8 * half) * NCOL + t * 16 + nl] = (h16)acc[t][r];
}

// ---------------------------------------------------------------------------

extern "C" void kernel_launch(void* const* d_in, const int* in_sizes, int n_in,
                              void* d_out, int out_size, void* d_ws, size_t ws_size,
                              hipStream_t stream) {
  const float* XQ = (const float*)d_in[0];
  const float* XK = (const float*)d_in[1];
  const float* XV = (const float*)d_in[2];
  const float* WQc = (const float*)d_in[3];
  const float* WKc = (const float*)d_in[4];
  const float* WVc = (const float*)d_in[5];
  const float* WQ = (const float*)d_in[6];
  const float* WK = (const float*)d_in[7];
  const float* WV = (const float*)d_in[8];
  const float* WO = (const float*)d_in[9];
  const float* WOb = (const float*)d_in[10];
  const int* mask = (const int*)d_in[11];
  float* out = (float*)d_out;

  char* ws = (char*)d_ws;
  size_t off = 0;
  auto alloc = [&](size_t bytes) -> void* {
    void* p = ws + off;
    off += (bytes + 255) & ~(size_t)255;
    return p;
  };
  const size_t XB = (size_t)MROWS * NCOL * sizeof(h16);  // 8 MB
  const size_t WB = (size_t)D_ * NCOL * sizeof(h16);     // 2 MB

  h16* Xq_h = (h16*)alloc(XB);
  h16* Xk_h = (h16*)alloc(XB);
  h16* Xv_h = (h16*)alloc(XB);
  h16* CWqt = (h16*)alloc(WB);
  h16* CWkt = (h16*)alloc(WB);
  h16* CWvt = (h16*)alloc(WB);
  h16* WOt  = (h16*)alloc(WB);
  h16* Qh   = (h16*)alloc(XB);
  h16* Kh   = (h16*)alloc(XB);
  h16* Vh   = (h16*)alloc(XB);
  h16* Vt   = (h16*)alloc(XB);
  h16* Hd   = (h16*)alloc(XB);
  float* mS = (float*)alloc((size_t)B_ * H_ * S_ * sizeof(float));
  float* zS = (float*)alloc((size_t)B_ * H_ * S_ * sizeof(float));
  unsigned char* flags = (unsigned char*)alloc(B_ * S_);

  const int T = 256;
  const int NX = MROWS * NCOL;

  k_f32_to_f16<<<(NX + T - 1) / T, T, 0, stream>>>(XQ, Xq_h, NX);
  k_f32_to_f16<<<(NX + T - 1) / T, T, 0, stream>>>(XK, Xk_h, NX);
  k_f32_to_f16<<<(NX + T - 1) / T, T, 0, stream>>>(XV, Xv_h, NX);
  k_combine_w<<<(D_ * NCOL + T - 1) / T, T, 0, stream>>>(WQc, WQ, CWqt);
  k_combine_w<<<(D_ * NCOL + T - 1) / T, T, 0, stream>>>(WKc, WK, CWkt);
  k_combine_w<<<(D_ * NCOL + T - 1) / T, T, 0, stream>>>(WVc, WV, CWvt);
  k_transpose_w<<<(D_ * NCOL + T - 1) / T, T, 0, stream>>>(WO, WOt);
  k_zero_flags<<<(B_ * S_ + T - 1) / T, T, 0, stream>>>(flags, B_ * S_);
  k_set_flags<<<(B_ * NM_ + T - 1) / T, T, 0, stream>>>(mask, flags);

  // fused projections: Q/K/V = X @ (Wcomb @ W[h])  -> f16 [B*S][H*64]
  dim3 gGemm(MROWS / 256, NCOL / 64);
  k_gemm16<<<gGemm, T, 0, stream>>>(Xq_h, CWqt, nullptr, nullptr, Qh, MROWS, NCOL, D_);
  k_gemm16<<<gGemm, T, 0, stream>>>(Xk_h, CWkt, nullptr, nullptr, Kh, MROWS, NCOL, D_);
  k_gemm16<<<gGemm, T, 0, stream>>>(Xv_h, CWvt, nullptr, nullptr, Vh, MROWS, NCOL, D_);
  k_transpose_v<<<(NX + T - 1) / T, T, 0, stream>>>(Vh, Vt);

  // attention: column-softmax stats then P@V
  dim3 gAttn(B_ * H_, S_ / 128);  // 8 waves/block, one 16-wide/high tile per wave
  k_attn_colstats<<<gAttn, T, 0, stream>>>(Qh, Kh, flags, mS, zS);
  k_attn_pv<<<gAttn, T, 0, stream>>>(Qh, Kh, Vt, mS, zS, flags, Hd);

  // output projection: out = heads @ WO + bias (f32)
  k_gemm16<<<gGemm, T, 0, stream>>>(Hd, WOt, WOb, out, nullptr, MROWS, D_, NCOL);

  (void)in_sizes; (void)n_in; (void)out_size; (void)ws_size;
}
